// GraphColorizer_67130338837052
// MI455X (gfx1250) — matrix operand order repaired and checked by reading, not compiled, not run on hardware
//
#include <hip/hip_runtime.h>

typedef unsigned short u16;
typedef __attribute__((ext_vector_type(16))) __bf16 v16bf;
typedef __attribute__((ext_vector_type(8)))  float  v8f;

#define N_V 3072
#define E_D 512
#define NH  8
#define HD  64
#define NC1 257   // NC+1

// ---------- helpers ----------
__device__ inline u16 f2bfu(float f) {
  union { float f; unsigned u; } a; a.f = f;
  unsigned r = a.u + 0x7FFFu + ((a.u >> 16) & 1u);
  return (u16)(r >> 16);
}
__device__ inline __bf16 bfu(u16 s) {
  union { u16 s; __bf16 b; } x; x.s = s; return x.b;
}
__device__ inline v8f v8zero() { v8f z; for (int i = 0; i < 8; ++i) z[i] = 0.f; return z; }

// ---------- cast f32 -> bf16 ----------
__global__ void cast_kernel(u16* __restrict__ dst, const float* __restrict__ src, int n) {
  int i = blockIdx.x * 256 + threadIdx.x;
  if (i < n) dst[i] = f2bfu(src[i]);
}

// ---------- WMMA GEMM with register double-buffered staging ----------
// C = act(A[MxK]*B[KxN] + bias) + resid ; A,B bf16 row-major; M%128==0, N%64==0, K%32==0.
// LDS tiles stored in *fragment order* -> each fragment = one 32B LDS read per lane.
__device__ inline void gemm_loadA(u16 (&av)[16], const u16* __restrict__ A,
                                  int row0, int K, int k0, int tid) {
#pragma unroll
  for (int i = 0; i < 16; ++i) {
    int x = tid + 256 * i, r = x >> 5, kk = x & 31;
    av[i] = A[(long)(row0 + r) * K + k0 + kk];
  }
}
__device__ inline void gemm_storeA(const u16 (&av)[16], u16* As, int tid) {
#pragma unroll
  for (int i = 0; i < 16; ++i) {
    int x = tid + 256 * i, r = x >> 5, kk = x & 31;
    int ln = (r & 15) + ((kk >> 3) & 1) * 16;
    int e  = (kk & 7) + ((kk >> 4) << 3);
    As[(r >> 4) * 512 + ln * 16 + e] = av[i];
  }
}
__device__ inline void gemm_loadB(u16 (&bv)[8], const u16* __restrict__ B,
                                  int col0, int N, int k0, int tid) {
#pragma unroll
  for (int i = 0; i < 8; ++i) {
    int x = tid + 256 * i, kk = x >> 6, n = x & 63;
    bv[i] = B[(long)(k0 + kk) * N + col0 + n];
  }
}
__device__ inline void gemm_storeB(const u16 (&bv)[8], u16* Bs, int tid) {
#pragma unroll
  for (int i = 0; i < 8; ++i) {
    int x = tid + 256 * i, kk = x >> 6, n = x & 63;
    int ln = (n & 15) + ((kk >> 3) & 1) * 16;
    Bs[(n >> 4) * 512 + ln * 16 + ((kk & 7) + ((kk >> 4) << 3))] = bv[i];
  }
}

__global__ __launch_bounds__(256) void wmma_gemm(
    const u16* __restrict__ A, const u16* __restrict__ B,
    float* outF, u16* outB, const float* bias, const float* resid,
    int M, int N, int K, int ldo, int coloff, int act)
{
  __shared__ __attribute__((aligned(32))) u16 As[2][8 * 32 * 16];  // double-buffered
  __shared__ __attribute__((aligned(32))) u16 Bs[2][4 * 32 * 16];
  int tid = threadIdx.x, lane = tid & 31, wv = tid >> 5;
  int row0 = blockIdx.x * 128, col0 = blockIdx.y * 64;
  v8f acc[4]; for (int c = 0; c < 4; ++c) acc[c] = v8zero();

  u16 av[16], bv[8];
  gemm_loadA(av, A, row0, K, 0, tid);
  gemm_loadB(bv, B, col0, N, 0, tid);
  gemm_storeA(av, &As[0][0], tid);
  gemm_storeB(bv, &Bs[0][0], tid);
  int buf = 0;

  for (int k0 = 0; k0 < K; k0 += 32) {
    __syncthreads();
    bool more = (k0 + 32) < K;
    if (more) {                       // next tile -> registers (overlaps WMMA chain)
      gemm_loadA(av, A, row0, K, k0 + 32, tid);
      gemm_loadB(bv, B, col0, N, k0 + 32, tid);
    }
    v16bf a  = *(const v16bf*)&As[buf][wv * 512 + lane * 16];
    v16bf b0 = *(const v16bf*)&Bs[buf][0 * 512 + lane * 16];
    v16bf b1 = *(const v16bf*)&Bs[buf][1 * 512 + lane * 16];
    v16bf b2 = *(const v16bf*)&Bs[buf][2 * 512 + lane * 16];
    v16bf b3 = *(const v16bf*)&Bs[buf][3 * 512 + lane * 16];
    acc[0] = __builtin_amdgcn_wmma_f32_16x16x32_bf16(false, a, false, b0, (short)0, acc[0], false, false);
    acc[1] = __builtin_amdgcn_wmma_f32_16x16x32_bf16(false, a, false, b1, (short)0, acc[1], false, false);
    acc[2] = __builtin_amdgcn_wmma_f32_16x16x32_bf16(false, a, false, b2, (short)0, acc[2], false, false);
    acc[3] = __builtin_amdgcn_wmma_f32_16x16x32_bf16(false, a, false, b3, (short)0, acc[3], false, false);
    if (more) {                       // stage into the other buffer
      gemm_storeA(av, &As[buf ^ 1][0], tid);
      gemm_storeB(bv, &Bs[buf ^ 1][0], tid);
    }
    buf ^= 1;
  }
  int half = lane >> 4, nn = lane & 15;
  for (int ch = 0; ch < 4; ++ch) {
    int coll = col0 + ch * 16 + nn;
    for (int e = 0; e < 8; ++e) {
      int row = row0 + wv * 16 + e + half * 8;
      float v = acc[ch][e];
      if (bias) v += bias[coll];
      if (act == 1)      v = v > 0.f ? v : 0.f;
      else if (act == 2) v = v > 0.f ? v : 0.05f * v;
      long idx = (long)row * ldo + coloff + coll;
      if (resid) v += resid[idx];
      if (outF) outF[idx] = v;
      if (outB) outB[idx] = f2bfu(v);
    }
  }
}

// ---------- masked flash attention (one head per blockIdx.y, 128 rows per WG) ----------
// mode 0: mask = adj>0 ; mode 1: mask = (adj==0 && r!=c)
__global__ __launch_bounds__(256) void flash_attn(
    const u16* __restrict__ Qb, const u16* __restrict__ Kb, const u16* __restrict__ Vb,
    const float* __restrict__ adj, u16* __restrict__ outB, int mode)
{
  __shared__ __attribute__((aligned(32))) u16 Ks[4][32 * 16];  // [subtile*2+kchunk] B-frags of K^T
  __shared__ __attribute__((aligned(32))) u16 Vs[4][32 * 16];  // [dchunk] B-frags of V
  __shared__ __attribute__((aligned(32))) u16 Ps[8][32 * 16];  // per-wave P in A-frag order
  __shared__ float adjS[128 * 32];
  int tid = threadIdx.x, lane = tid & 31, wv = tid >> 5;
  int row0 = blockIdx.x * 128, h = blockIdx.y;
  int half = lane >> 4, nn = lane & 15;

  // Q fragments (16 rows x 64 dims = two 16x32 A-frags), gathered once
  v16bf aQ[2];
  int qrow = row0 + wv * 16 + nn;
  for (int c = 0; c < 2; ++c)
    for (int e = 0; e < 16; ++e) {
      int k = e + (half << 4) + c * 32;
      aQ[c][e] = bfu(Qb[(long)qrow * E_D + h * HD + k]);
    }

  float mrun[8], lrun[8];
  v8f accO[4];
  for (int e = 0; e < 8; ++e) { mrun[e] = -1e30f; lrun[e] = 0.f; }
  for (int c = 0; c < 4; ++c) accO[c] = v8zero();

  for (int kb = 0; kb < N_V / 32; ++kb) {
    int c0 = kb * 32;
    __syncthreads();
    for (int x = tid; x < 32 * 64; x += 256) {        // K^T B-frags
      int kk = x >> 6, d = x & 63;
      u16 val = Kb[(long)(c0 + kk) * E_D + h * HD + d];
      int sub = kk >> 4, k = d & 31;
      int ln = (kk & 15) + ((k >> 4) & 1) * 16;
      Ks[sub * 2 + (d >> 5)][ln * 16 + (k & 15)] = val;
    }
    for (int x = tid; x < 32 * 64; x += 256) {        // V B-frags
      int kk = x >> 6, d = x & 63;
      u16 val = Vb[(long)(c0 + kk) * E_D + h * HD + d];
      int ln = (d & 15) + ((kk >> 3) & 1) * 16;
      Vs[d >> 4][ln * 16 + ((kk & 7) + ((kk >> 4) << 3))] = val;
    }
    for (int x = tid; x < 128 * 32; x += 256) {       // adjacency tile
      int r = x >> 5, cc = x & 31;
      adjS[r * 32 + cc] = adj[(long)(row0 + r) * N_V + c0 + cc];
    }
    __syncthreads();

    // prefetch next block's K/V slabs and adjacency slab (global_prefetch_b8),
    // overlapping with the WMMA/softmax work below
    if (kb + 1 < N_V / 32) {
      int c1 = c0 + 32;
      if (tid < 32) {
        __builtin_prefetch(&Kb[(long)(c1 + tid) * E_D + h * HD], 0, 1);
        __builtin_prefetch(&Vb[(long)(c1 + tid) * E_D + h * HD], 0, 1);
      } else if (tid < 160) {
        int r = tid - 32;
        __builtin_prefetch(&adj[(long)(row0 + r) * N_V + c1], 0, 1);
      }
    }

    float p[2][8];
    for (int sub = 0; sub < 2; ++sub) {
      v8f s = v8zero();
      v16bf b0 = *(const v16bf*)&Ks[sub * 2 + 0][lane * 16];
      v16bf b1 = *(const v16bf*)&Ks[sub * 2 + 1][lane * 16];
      s = __builtin_amdgcn_wmma_f32_16x16x32_bf16(false, aQ[0], false, b0, (short)0, s, false, false);
      s = __builtin_amdgcn_wmma_f32_16x16x32_bf16(false, aQ[1], false, b1, (short)0, s, false, false);
      for (int e = 0; e < 8; ++e) {
        int r = wv * 16 + e + half * 8;
        int cc = sub * 16 + nn;
        float a = adjS[r * 32 + cc];
        bool valid = (mode == 0) ? (a > 0.f) : (a == 0.f && (row0 + r) != (c0 + cc));
        p[sub][e] = valid ? s[e] * 0.125f : -1e9f;
      }
    }
    // online softmax per row (16 columns live across one 16-lane half)
    for (int e = 0; e < 8; ++e) {
      float m = fmaxf(p[0][e], p[1][e]);
      for (int off = 8; off > 0; off >>= 1) m = fmaxf(m, __shfl_xor(m, off, 32));
      float nm = fmaxf(mrun[e], m);
      float alpha = __expf(mrun[e] - nm);
      float e0 = __expf(p[0][e] - nm), e1 = __expf(p[1][e] - nm);
      float rs = e0 + e1;
      for (int off = 8; off > 0; off >>= 1) rs += __shfl_xor(rs, off, 32);
      lrun[e] = lrun[e] * alpha + rs;
      mrun[e] = nm;
      for (int c = 0; c < 4; ++c) accO[c][e] *= alpha;
      p[0][e] = e0; p[1][e] = e1;
    }
    // transpose P (C-layout) -> A-frag order via per-wave LDS scratch
    for (int sub = 0; sub < 2; ++sub)
      for (int e = 0; e < 8; ++e) {
        int m = e + half * 8, k = sub * 16 + nn;
        int ln = m + ((k >> 3) & 1) * 16;
        Ps[wv][ln * 16 + ((k & 7) + ((k >> 4) << 3))] = f2bfu(p[sub][e]);
      }
    __syncthreads();
    v16bf pa = *(const v16bf*)&Ps[wv][lane * 16];
    for (int c = 0; c < 4; ++c) {
      v16bf b = *(const v16bf*)&Vs[c][lane * 16];
      accO[c] = __builtin_amdgcn_wmma_f32_16x16x32_bf16(false, pa, false, b, (short)0, accO[c], false, false);
    }
  }
  for (int c = 0; c < 4; ++c)
    for (int e = 0; e < 8; ++e) {
      int row = row0 + wv * 16 + e + half * 8;
      int col = h * HD + c * 16 + nn;
      outB[(long)row * E_D + col] = f2bfu(accO[c][e] / lrun[e]);
    }
}

// ---------- small one-time matmuls (f32 VALU) ----------
__global__ void simple_mm(float* __restrict__ C, const float* __restrict__ A,
                          const float* __restrict__ B, int M, int Nc, int K, int transB) {
  int id = blockIdx.x * 256 + threadIdx.x;
  if (id >= M * Nc) return;
  int m = id / Nc, n = id % Nc;
  float a = 0.f;
  if (transB) { for (int k = 0; k < K; ++k) a += A[(long)m * K + k] * B[(long)n * K + k]; }
  else        { for (int k = 0; k < K; ++k) a += A[(long)m * K + k] * B[(long)k * Nc + n]; }
  C[(long)m * Nc + n] = a;
}

// ---------- degrees / sort / mean / qg ----------
__global__ void degree_kernel(const float* __restrict__ adj, unsigned* __restrict__ keys) {
  __shared__ float r[256];
  int row = blockIdx.x; float s = 0.f;
  for (int c = threadIdx.x; c < N_V; c += 256) s += adj[(long)row * N_V + c];
  r[threadIdx.x] = s; __syncthreads();
  for (int st = 128; st > 0; st >>= 1) { if (threadIdx.x < st) r[threadIdx.x] += r[threadIdx.x + st]; __syncthreads(); }
  if (threadIdx.x == 0) {
    unsigned deg = (unsigned)(r[0] + 0.5f);
    keys[row] = (deg << 12) | (4095u - (unsigned)row);  // stable argsort(-deg)
  }
}

__global__ __launch_bounds__(1024) void sort_kernel(const unsigned* __restrict__ keys, int* __restrict__ order) {
  __shared__ unsigned s[4096];
  int t = threadIdx.x;
  for (int i = t; i < 4096; i += 1024) s[i] = (i < N_V) ? keys[i] : 0u;
  __syncthreads();
  for (int k = 2; k <= 4096; k <<= 1)
    for (int j = k >> 1; j > 0; j >>= 1) {
      for (int i = t; i < 4096; i += 1024) {
        int ixj = i ^ j;
        if (ixj > i) {
          unsigned a = s[i], b = s[ixj];
          bool up = ((i & k) == 0);
          bool sw = up ? (a < b) : (a > b);   // descending overall
          if (sw) { s[i] = b; s[ixj] = a; }
        }
      }
      __syncthreads();
    }
  for (int i = t; i < N_V; i += 1024) order[i] = 4095 - (int)(s[i] & 0xFFFu);
}

__global__ void colmean_kernel(const float* __restrict__ xf, float* __restrict__ g) {
  __shared__ float r[256];
  int col = blockIdx.x; float s = 0.f;
  for (int row = threadIdx.x; row < N_V; row += 256) s += xf[(long)row * E_D + col];
  r[threadIdx.x] = s; __syncthreads();
  for (int st = 128; st > 0; st >>= 1) { if (threadIdx.x < st) r[threadIdx.x] += r[threadIdx.x + st]; __syncthreads(); }
  if (threadIdx.x == 0) g[col] = r[0] / (float)N_V;
}

__global__ void qg_kernel(const float* __restrict__ g, const float* __restrict__ Wbot, float* __restrict__ qg) {
  int j = threadIdx.x;
  if (j < E_D) { float a = 0.f; for (int i = 0; i < E_D; ++i) a += g[i] * Wbot[(long)i * E_D + j]; qg[j] = a; }
}

// ---------- persistent sequential coloring (1 WG, 32 waves) ----------
__global__ __launch_bounds__(1024) void seq_kernel(
    float* __restrict__ out, const int* __restrict__ order,
    const float* __restrict__ Qall, const float* __restrict__ Kc,
    const float* __restrict__ Vc, const float* __restrict__ PptT,
    const float* __restrict__ adj, const float* __restrict__ baseline)
{
  __shared__ int colorsS[N_V];
  __shared__ int bins[NC1];
  __shared__ float qv[E_D];
  __shared__ float scS[NH * NC1];
  __shared__ float red[1024];
  __shared__ float mh[NH], sh[NH];
  __shared__ float attnC[E_D];
  __shared__ float psS[512];
  __shared__ int rawS, nUsedS; __shared__ float logpS;
  int tid = threadIdx.x;

  for (int i = tid; i < N_V; i += 1024) colorsS[i] = -1;
  if (tid == 0) { nUsedS = 1; logpS = 0.f; }
  __syncthreads();
  if (tid == 0) colorsS[order[0]] = 0;
  __syncthreads();

  for (int t = 1; t < N_V; ++t) {
    int v = order[t];
    for (int c = tid; c < NC1; c += 1024) bins[c] = 0;
    __syncthreads();
    const float* arow = adj + (long)v * N_V;
    for (int w = tid; w < N_V; w += 1024) {
      if (arow[w] > 0.f) { int c = colorsS[w]; if (c >= 0 && c < NC1) atomicOr(&bins[c], 1); }
    }
    if (tid < E_D) qv[tid] = Qall[(long)v * E_D + tid];
    __syncthreads();
    // context-attention scores (8 heads x 257 colors)
    for (int p = tid; p < NH * NC1; p += 1024) {
      int c = p >> 3, h = p & 7;
      const float* kc = Kc + (long)c * E_D + h * HD;
      const float* q = qv + h * HD;
      float a = 0.f;
      for (int d = 0; d < HD; ++d) a += q[d] * kc[d];
      scS[h * NC1 + c] = a * 0.125f;
    }
    __syncthreads();
    // per-head softmax (8 groups of 128 threads)
    int grp = tid >> 7, l = tid & 127;
    float pm = -1e30f;
    for (int c = l; c < NC1; c += 128) pm = fmaxf(pm, scS[grp * NC1 + c]);
    red[tid] = pm; __syncthreads();
    for (int s = 64; s > 0; s >>= 1) { if (l < s) red[tid] = fmaxf(red[tid], red[tid + s]); __syncthreads(); }
    if (l == 0) mh[grp] = red[tid];
    __syncthreads();
    float psum = 0.f;
    for (int c = l; c < NC1; c += 128) { float e = __expf(scS[grp * NC1 + c] - mh[grp]); scS[grp * NC1 + c] = e; psum += e; }
    red[tid] = psum; __syncthreads();
    for (int s = 64; s > 0; s >>= 1) { if (l < s) red[tid] += red[tid + s]; __syncthreads(); }
    if (l == 0) sh[grp] = red[tid];
    __syncthreads();
    // combine with Vc
    if (tid < E_D) {
      int h = tid >> 6;
      const float* e = scS + h * NC1;
      float a = 0.f;
      for (int c = 0; c < NC1; ++c) a += e[c] * Vc[(long)c * E_D + tid];
      attnC[tid] = a / sh[h];
    }
    __syncthreads();
    // pointer scores via fused PptT
    for (int c = tid; c < NC1; c += 1024) {
      float a = 0.f;
      for (int i = 0; i < E_D; ++i) a += attnC[i] * PptT[(long)i * NC1 + c];
      psS[c] = a * 0.04419417382f;   // 1/sqrt(512)
    }
    __syncthreads();
    int nu = nUsedS;
    if (tid < 512) {
      float val;
      if (tid < NC1) {
        bool forb = (bins[tid] != 0) || ((tid >= nu) && (tid != 256));
        val = forb ? -1e9f : psS[tid];
      } else val = -2e9f;
      psS[tid] = val; red[tid] = val;
    }
    __syncthreads();
    for (int s = 256; s > 0; s >>= 1) { if (tid < s) red[tid] = fmaxf(red[tid], red[tid + s]); __syncthreads(); }
    float mx = red[0];
    if (tid == 0) rawS = 0x7fffffff;
    __syncthreads();
    if (tid < NC1 && psS[tid] == mx) atomicMin(&rawS, tid);
    if (tid < 512) red[tid] = (tid < NC1) ? __expf(psS[tid] - mx) : 0.f;
    __syncthreads();
    for (int s = 256; s > 0; s >>= 1) { if (tid < s) red[tid] += red[tid + s]; __syncthreads(); }
    if (tid == 0) {
      float sum = red[0];
      int raw = rawS;
      float pr = 1.0f / sum;            // probs[argmax] = exp(0)/sum
      logpS += logf(pr + 1e-8f) - logf(1e-8f);
      int isnew = (raw == 256) ? 1 : 0;
      colorsS[v] = isnew ? nUsedS : raw;
      nUsedS += isnew;
    }
    __syncthreads();
  }
  for (int i = tid; i < N_V; i += 1024) out[i] = (float)colorsS[i];
  if (tid == 0) out[N_V] = ((float)nUsedS - baseline[0]) * logpS / (float)N_V;
}

// ---------- host orchestration ----------
extern "C" void kernel_launch(void* const* d_in, const int* in_sizes, int n_in,
                              void* d_out, int out_size, void* d_ws, size_t ws_size,
                              hipStream_t stream) {
  const int N = N_V, E = E_D;
  const float* emb  = (const float*)d_in[0];
  const float* adj  = (const float*)d_in[1];
  const float* basl = (const float*)d_in[2];
  const float* nW   = (const float*)d_in[3];
  const float* mW   = (const float*)d_in[4];
  const float* cW1  = (const float*)d_in[5];
  const float* cb1  = (const float*)d_in[6];
  const float* cW2  = (const float*)d_in[7];
  const float* cb2  = (const float*)d_in[8];
  const float* cuWq = (const float*)d_in[9];
  const float* cuWk = (const float*)d_in[10];
  const float* cuWv = (const float*)d_in[11];
  const float* cuWo = (const float*)d_in[12];
  const float* pWq  = (const float*)d_in[13];
  const float* pWk  = (const float*)d_in[14];
  const float* ctab = (const float*)d_in[15];
  float* out = (float*)d_out;

  // bump allocator over workspace
  char* p = (char*)d_ws;
  auto alloc = [&](size_t bytes) { char* r = p; p += (bytes + 255) & ~(size_t)255; return (void*)r; };
  float* xf   = (float*)alloc((size_t)N * E * 4);
  u16* xb     = (u16*)alloc((size_t)N * E * 2);
  u16* qb     = (u16*)alloc((size_t)N * E * 2);
  u16* kb     = (u16*)alloc((size_t)N * E * 2);
  u16* vb     = (u16*)alloc((size_t)N * E * 2);
  u16* attnb  = (u16*)alloc((size_t)N * E * 2);
  u16* catb   = (u16*)alloc((size_t)N * 2 * E * 2);
  u16* h1b    = (u16*)alloc((size_t)N * 2 * E * 2);
  u16* nWb    = (u16*)alloc((size_t)3 * 4 * E * E * 2);
  u16* mWb    = (u16*)alloc((size_t)3 * 4 * E * E * 2);
  u16* cW1b   = (u16*)alloc((size_t)3 * 2 * E * 2 * E * 2);
  u16* cW2b   = (u16*)alloc((size_t)3 * 2 * E * E * 2);
  u16* cuWqTB = (u16*)alloc((size_t)E * E * 2);
  float* g    = (float*)alloc(E * 4);
  float* qg   = (float*)alloc(E * 4);
  float* Qall = (float*)alloc((size_t)N * E * 4);
  float* Kc   = (float*)alloc((size_t)NC1 * E * 4);
  float* Vc   = (float*)alloc((size_t)NC1 * E * 4);
  float* Kp   = (float*)alloc((size_t)NC1 * E * 4);
  float* M1   = (float*)alloc((size_t)E * E * 4);
  float* PptT = (float*)alloc((size_t)E * NC1 * 4);
  unsigned* keys = (unsigned*)alloc((size_t)N * 4);
  int* order  = (int*)alloc((size_t)N * 4);

  auto cast = [&](u16* d, const float* s, int n) {
    cast_kernel<<<(n + 255) / 256, 256, 0, stream>>>(d, s, n);
  };
  cast(nWb,  nW,  3 * 4 * E * E);
  cast(mWb,  mW,  3 * 4 * E * E);
  cast(cW1b, cW1, 3 * 2 * E * 2 * E);
  cast(cW2b, cW2, 3 * 2 * E * E);
  cast(cuWqTB, cuWq, E * E);          // top half of cuWq (rows 0..E-1)
  hipMemcpyAsync(xf, emb, (size_t)N * E * 4, hipMemcpyDeviceToDevice, stream);
  cast(xb, emb, N * E);

  dim3 blk(256);
  for (int l = 0; l < 3; ++l) {
    const u16* WN = nWb + (size_t)l * 4 * E * E;
    const u16* WM = mWb + (size_t)l * 4 * E * E;
    // neighbor branch
    wmma_gemm<<<dim3(N / 128, E / 64), blk, 0, stream>>>(xb, WN + 0 * E * E, nullptr, qb, nullptr, nullptr, N, E, E, E, 0, 0);
    wmma_gemm<<<dim3(N / 128, E / 64), blk, 0, stream>>>(xb, WN + 1 * E * E, nullptr, kb, nullptr, nullptr, N, E, E, E, 0, 0);
    wmma_gemm<<<dim3(N / 128, E / 64), blk, 0, stream>>>(xb, WN + 2 * E * E, nullptr, vb, nullptr, nullptr, N, E, E, E, 0, 0);
    flash_attn<<<dim3(N / 128, NH), blk, 0, stream>>>(qb, kb, vb, adj, attnb, 0);
    wmma_gemm<<<dim3(N / 128, E / 64), blk, 0, stream>>>(attnb, WN + 3 * E * E, nullptr, catb, nullptr, nullptr, N, E, E, 2 * E, 0, 1);
    // non-neighbor branch
    wmma_gemm<<<dim3(N / 128, E / 64), blk, 0, stream>>>(xb, WM + 0 * E * E, nullptr, qb, nullptr, nullptr, N, E, E, E, 0, 0);
    wmma_gemm<<<dim3(N / 128, E / 64), blk, 0, stream>>>(xb, WM + 1 * E * E, nullptr, kb, nullptr, nullptr, N, E, E, E, 0, 0);
    wmma_gemm<<<dim3(N / 128, E / 64), blk, 0, stream>>>(xb, WM + 2 * E * E, nullptr, vb, nullptr, nullptr, N, E, E, E, 0, 0);
    flash_attn<<<dim3(N / 128, NH), blk, 0, stream>>>(qb, kb, vb, adj, attnb, 1);
    wmma_gemm<<<dim3(N / 128, E / 64), blk, 0, stream>>>(attnb, WM + 3 * E * E, nullptr, catb, nullptr, nullptr, N, E, E, 2 * E, E, 1);
    // MLP + residual
    wmma_gemm<<<dim3(N / 128, 2 * E / 64), blk, 0, stream>>>(catb, cW1b + (size_t)l * 2 * E * 2 * E, nullptr, h1b,
                                                             cb1 + (size_t)l * 2 * E, nullptr, N, 2 * E, 2 * E, 2 * E, 0, 2);
    wmma_gemm<<<dim3(N / 128, E / 64), blk, 0, stream>>>(h1b, cW2b + (size_t)l * 2 * E * E, xf, xb,
                                                         cb2 + (size_t)l * E, xf, N, E, 2 * E, E, 0, 0);
  }

  // scan precompute
  colmean_kernel<<<E, 256, 0, stream>>>(xf, g);
  qg_kernel<<<1, E, 0, stream>>>(g, cuWq + (size_t)E * E, qg);  // bottom half of cuWq
  wmma_gemm<<<dim3(N / 128, E / 64), blk, 0, stream>>>(xb, cuWqTB, Qall, nullptr, qg, nullptr, N, E, E, E, 0, 0);
  simple_mm<<<(NC1 * E + 255) / 256, 256, 0, stream>>>(Kc, ctab, cuWk, NC1, E, E, 0);
  simple_mm<<<(NC1 * E + 255) / 256, 256, 0, stream>>>(Vc, ctab, cuWv, NC1, E, E, 0);
  simple_mm<<<(NC1 * E + 255) / 256, 256, 0, stream>>>(Kp, ctab, pWk, NC1, E, E, 0);
  simple_mm<<<(E * E + 255) / 256, 256, 0, stream>>>(M1, cuWo, pWq, E, E, E, 0);
  simple_mm<<<(E * NC1 + 255) / 256, 256, 0, stream>>>(PptT, M1, Kp, E, NC1, E, 1); // PptT[i][c] = M1[i]·Kp[c]
  degree_kernel<<<N, 256, 0, stream>>>(adj, keys);
  sort_kernel<<<1, 1024, 0, stream>>>(keys, order);

  // sequential greedy coloring
  seq_kernel<<<1, 1024, 0, stream>>>(out, order, Qall, Kc, Vc, PptT, adj, basl);
}
